// QLSTM_65481071408547
// MI455X (gfx1250) — matrix operand order repaired and checked by reading, not compiled, not run on hardware
//
#include <hip/hip_runtime.h>

// LSTM: T=512, B=64, D=512, H=512.  Persistent kernel, one grid barrier/step.
// GEMM per step: [64 x 1024] @ [1024 x 2048] via v_wmma_f32_16x16x32_f16,
// weights resident in LDS (f16, pre-swizzled into WMMA B fragments).
// Recurrent h kept in a double-buffered f16 side buffer (converted once at
// write time) so the hot loop's h-half loads are pure b128 loads, no cvt.

#define T_  512
#define B_  64
#define D_  512
#define H_  512
#define K_  (D_ + H_)   // 1024
#define NWG 32          // one workgroup per 16-wide hidden-column tile
#define BLOCK 128       // 4 waves = 4 batch m-tiles

typedef _Float16 h16;
typedef __attribute__((ext_vector_type(16))) _Float16 v16h;
typedef __attribute__((ext_vector_type(8)))  _Float16 v8h;
typedef __attribute__((ext_vector_type(8)))  float    v8f;
typedef __attribute__((ext_vector_type(4)))  float    v4f;

__device__ __forceinline__ float fsigmoid(float x) {
  return 1.0f / (1.0f + __expf(-x));
}
__device__ __forceinline__ float ftanh_(float x) {
  return 2.0f / (1.0f + __expf(-2.0f * x)) - 1.0f;
}

// A fragment (16-bit 16x32, ISA 7.12.2): this lane holds row M=lane&15,
// K = koff+{0..7} and koff+{16..23}.  f32 source: load + convert in-register.
__device__ __forceinline__ v16h load_a_f32cvt(const float* p) {
  v4f u0 = *(const v4f*)(p);
  v4f u1 = *(const v4f*)(p + 4);
  v4f u2 = *(const v4f*)(p + 16);
  v4f u3 = *(const v4f*)(p + 20);
  v16h a;
#pragma unroll
  for (int c = 0; c < 4; ++c) {
    a[c]      = (h16)u0[c];
    a[c + 4]  = (h16)u1[c];
    a[c + 8]  = (h16)u2[c];
    a[c + 12] = (h16)u3[c];
  }
  return a;
}

// A fragment from an f16 source: two b128 loads, no conversion.
__device__ __forceinline__ v16h load_a_f16(const h16* p) {
  v8h a0 = *(const v8h*)(p);       // K koff+0..7
  v8h a1 = *(const v8h*)(p + 16);  // K koff+16..23
  return __builtin_shufflevector(a0, a1,
      0,1,2,3,4,5,6,7,8,9,10,11,12,13,14,15);
}

// B fragment: 16 consecutive f16 staged in LDS (32 bytes) -> two ds_load_b128.
__device__ __forceinline__ v16h load_b_lds(const h16* p) {
  v8h b0 = *(const v8h*)(p);
  v8h b1 = *(const v8h*)(p + 8);
  return __builtin_shufflevector(b0, b1,
      0,1,2,3,4,5,6,7,8,9,10,11,12,13,14,15);
}

// Monotonic-counter grid barrier (all NWG workgroups resident by construction).
__device__ __forceinline__ void grid_barrier(int* ctr, int idx) {
  __threadfence();
  __syncthreads();
  if (threadIdx.x == 0) {
    __hip_atomic_fetch_add(ctr, 1, __ATOMIC_RELEASE, __HIP_MEMORY_SCOPE_AGENT);
    while (__hip_atomic_load(ctr, __ATOMIC_ACQUIRE, __HIP_MEMORY_SCOPE_AGENT)
           < idx * NWG) {
      __builtin_amdgcn_s_sleep(2);
    }
  }
  __syncthreads();
}

__global__ __launch_bounds__(BLOCK, 1)
void lstm_persistent(const float* __restrict__ x,
                     const float* __restrict__ Wf, const float* __restrict__ bf,
                     const float* __restrict__ Wi, const float* __restrict__ bi,
                     const float* __restrict__ Wg, const float* __restrict__ bg,
                     const float* __restrict__ Wo, const float* __restrict__ bo,
                     float* __restrict__ out,   // [T,B,H] ++ hx[B,H] ++ cx[B,H]
                     h16* __restrict__ hbuf,    // [2][B][H] f16 recurrent state
                     int* __restrict__ ctr) {
  // f16 weights pre-swizzled as B fragments: [gate][kk][lane][16 halfs] = 128 KB
  __shared__ __attribute__((aligned(16))) h16 ldsW[4][32][32][16];

  const int tid  = threadIdx.x;
  const int lane = tid & 31;
  const int wave = tid >> 5;      // batch m-tile
  const int wg   = blockIdx.x;    // hidden n-tile

  const int n0   = wg * 16;
  const int m0   = wave * 16;
  const int row  = m0 + (lane & 15);        // A row (batch) for this lane
  const int nn   = n0 + (lane & 15);        // output column (hidden unit)
  const int koff = (lane < 16) ? 0 : 8;     // A-fragment K sub-offset

  // ---- one-time: stage f16 weights into LDS, swizzled for direct B loads ----
  for (int e = tid; e < 4 * 32 * 32; e += BLOCK) {
    const int g  = e >> 10;        // gate
    const int kk = (e >> 5) & 31;  // 32-wide K chunk
    const int ln = e & 31;         // destination lane
    const float* W = (g == 0) ? Wf : (g == 1) ? Wi : (g == 2) ? Wg : Wo;
    // B (32x16): lane = column N = W row; lane<16 -> K 0..7,16..23, else K 8..15,24..31
    const float* src = W + (size_t)(n0 + (ln & 15)) * K_
                         + kk * 32 + ((ln < 16) ? 0 : 8);
    h16* dst = &ldsW[g][kk][ln][0];
#pragma unroll
    for (int c = 0; c < 16; ++c) dst[c] = (h16)src[c + (c & 8)];
  }
  const float vbf = bf[nn], vbi = bi[nn], vbg = bg[nn], vbo = bo[nn];
  __syncthreads();

  v8f   cstate = {};    // cell state for this (m,n) tile, f32, lives 512 steps
  float hlast[8];

  for (int t = 0; t < T_; ++t) {
    v8f accF = {}, accI = {}, accG = {}, accO = {};

    // ---- x contribution: K = 0..511 ----
    const float* xrow = x + ((size_t)t * B_ + row) * D_;
#pragma unroll 4
    for (int kk = 0; kk < 16; ++kk) {
      v16h a  = load_a_f32cvt(xrow + kk * 32 + koff);
      v16h bF = load_b_lds(&ldsW[0][kk][lane][0]);
      v16h bI = load_b_lds(&ldsW[1][kk][lane][0]);
      v16h bG = load_b_lds(&ldsW[2][kk][lane][0]);
      v16h bO = load_b_lds(&ldsW[3][kk][lane][0]);
      accF = __builtin_amdgcn_wmma_f32_16x16x32_f16(false, a, false, bF, (short)0, accF, false, false);
      accI = __builtin_amdgcn_wmma_f32_16x16x32_f16(false, a, false, bI, (short)0, accI, false, false);
      accG = __builtin_amdgcn_wmma_f32_16x16x32_f16(false, a, false, bG, (short)0, accG, false, false);
      accO = __builtin_amdgcn_wmma_f32_16x16x32_f16(false, a, false, bO, (short)0, accO, false, false);
    }

    // ---- h contribution: K = 512..1023 ; f16 h_{t-1} in hbuf[(t-1)&1] ----
    if (t > 0) {
      const h16* hrow = hbuf + ((size_t)((t - 1) & 1) * B_ + row) * H_;
#pragma unroll 4
      for (int kk = 0; kk < 16; ++kk) {
        v16h a  = load_a_f16(hrow + kk * 32 + koff);
        v16h bF = load_b_lds(&ldsW[0][kk + 16][lane][0]);
        v16h bI = load_b_lds(&ldsW[1][kk + 16][lane][0]);
        v16h bG = load_b_lds(&ldsW[2][kk + 16][lane][0]);
        v16h bO = load_b_lds(&ldsW[3][kk + 16][lane][0]);
        accF = __builtin_amdgcn_wmma_f32_16x16x32_f16(false, a, false, bF, (short)0, accF, false, false);
        accI = __builtin_amdgcn_wmma_f32_16x16x32_f16(false, a, false, bI, (short)0, accI, false, false);
        accG = __builtin_amdgcn_wmma_f32_16x16x32_f16(false, a, false, bG, (short)0, accG, false, false);
        accO = __builtin_amdgcn_wmma_f32_16x16x32_f16(false, a, false, bO, (short)0, accO, false, false);
      }
    }

    // ---- gates + state update (C/D layout: lane<16 -> M=m0+j, else m0+8+j) ----
    h16* hwr = hbuf + (size_t)(t & 1) * B_ * H_;
#pragma unroll
    for (int j = 0; j < 8; ++j) {
      const float f = fsigmoid(accF[j] + vbf);
      const float i = fsigmoid(accI[j] + vbi);
      const float g = ftanh_  (accG[j] + vbg);
      const float o = fsigmoid(accO[j] + vbo);
      const float c = f * cstate[j] + i * g;
      cstate[j] = c;
      const float h = o * ftanh_(c);
      hlast[j] = h;
      const int m = m0 + j + ((lane < 16) ? 0 : 8);
      out[((size_t)t * B_ + m) * H_ + nn] = h;       // fp32 output
      hwr[(size_t)m * H_ + nn] = (h16)h;             // f16 recurrent state
    }

    if (t + 1 < T_) grid_barrier(ctr, t + 1);
  }

  // ---- final hx, cx ----
  float* hx = out + (size_t)T_ * B_ * H_;
  float* cx = hx + (size_t)B_ * H_;
#pragma unroll
  for (int j = 0; j < 8; ++j) {
    const int m = m0 + j + ((lane < 16) ? 0 : 8);
    hx[(size_t)m * H_ + nn] = hlast[j];
    cx[(size_t)m * H_ + nn] = cstate[j];
  }
}

extern "C" void kernel_launch(void* const* d_in, const int* in_sizes, int n_in,
                              void* d_out, int out_size, void* d_ws, size_t ws_size,
                              hipStream_t stream) {
  const float* x  = (const float*)d_in[0];
  const float* Wf = (const float*)d_in[1];
  const float* bf = (const float*)d_in[2];
  const float* Wi = (const float*)d_in[3];
  const float* bi = (const float*)d_in[4];
  const float* Wg = (const float*)d_in[5];
  const float* bg = (const float*)d_in[6];
  const float* Wo = (const float*)d_in[7];
  const float* bo = (const float*)d_in[8];
  float* out = (float*)d_out;
  int*   ctr = (int*)d_ws;
  h16*   hbuf = (h16*)((char*)d_ws + 1024);   // 2*B*H f16 = 128 KB

  // reset grid-barrier counter every launch (graph-capture-safe memset node)
  hipMemsetAsync(d_ws, 0, 64, stream);

  lstm_persistent<<<NWG, BLOCK, 0, stream>>>(x, Wf, bf, Wi, bi, Wg, bg, Wo, bo,
                                             out, hbuf, ctr);
}
